// m2oAttentionB2_32177894982195
// MI455X (gfx1250) — matrix operand order repaired and checked by reading, not compiled, not run on hardware
//
#include <hip/hip_runtime.h>

typedef __attribute__((ext_vector_type(16))) _Float16 v16h;
typedef __attribute__((ext_vector_type(8)))  _Float16 half8;
typedef __attribute__((ext_vector_type(4)))  _Float16 half4;
typedef __attribute__((ext_vector_type(8)))  float    v8f;

#define NB 64
#define NS 50
#define NT 200
#define ND 128
#define NH 64
#define TPAD 208      // T padded to 13 * 16
#define DP   136      // padded row length in halves (272 B -> rotates banks by 4/row)

// Branchless tanh on the fast HW transcendentals:
// tanh(x) = sign(x) * (1 - 2/(exp(2|x|)+1)); exp2 overflow -> inf -> correct +/-1.
__device__ __forceinline__ float fast_tanh(float x) {
    const float TWO_LOG2E = 2.885390081777927f;   // 2*log2(e)
    float a = __builtin_fabsf(x);
    float e = __builtin_amdgcn_exp2f(a * TWO_LOG2E);
    float r = __builtin_amdgcn_rcpf(e + 1.0f);
    float t = __builtin_fmaf(-2.0f, r, 1.0f);
    return __builtin_copysignf(t, x);
}

__device__ __forceinline__ float fast_exp(float x) {
    const float LOG2E = 1.4426950408889634f;
    return __builtin_amdgcn_exp2f(x * LOG2E);     // x <= 0 here; underflow -> 0 is fine
}

__launch_bounds__(256)
__global__ void m2o_att_kernel(const float* __restrict__ inp,
                               const float* __restrict__ z,
                               const int*   __restrict__ mask,
                               const float* __restrict__ W1,
                               const float* __restrict__ W2w,
                               const float* __restrict__ W2b,
                               const float* __restrict__ Vw,
                               const float* __restrict__ Vb,
                               float* __restrict__ out)
{
    __shared__ _Float16 sA[TPAD * DP];      // 56576 B: input tile, f16
    __shared__ float s_z[ND];               //   512 B
    __shared__ float s_w2h[NH];             //   256 B
    __shared__ float s_attp[4][TPAD];       //  3328 B: per-N-tile partial att / scratch
    __shared__ float s_att[TPAD];           //   832 B: softmax result

    const int tid = threadIdx.x;
    const int bs  = blockIdx.x;             // b*S + s
    const float* X = inp + (size_t)bs * (NT * ND);

    // ---- Phase 0: stage input tile (f32 -> f16) into LDS; 25 float4 per thread
    for (int i = tid; i < (NT * ND) / 4; i += 256) {
        float4 v = ((const float4*)X)[i];
        int t = i >> 5;                 // (i*4)/128
        int d = (i & 31) << 2;
        half4 hv = { (_Float16)v.x, (_Float16)v.y, (_Float16)v.z, (_Float16)v.w };
        *(half4*)(&sA[t * DP + d]) = hv;
    }
    // zero the 8 pad rows (t = 200..207) so the last M-tile is clean
    for (int i = tid; i < 8 * DP; i += 256) sA[NT * DP + i] = (_Float16)0.f;
    if (tid < ND) s_z[tid] = z[(size_t)bs * ND + tid];
    __syncthreads();

    // ---- Phase 1: w2h[h] = z . W2w[h] + W2b[h]   (tiny, 64 threads)
    if (tid < NH) {
        float acc = W2b[tid];
        const float* wr = W2w + tid * ND;
        for (int d = 0; d < ND; ++d) acc += s_z[d] * wr[d];
        s_w2h[tid] = acc;
    }

    const int lane  = tid & 31;
    const int wid   = tid >> 5;
    const int ncol  = wid & 3;              // N-tile (h block) this wave owns
    const int mgrp  = wid >> 2;             // M-tile parity
    const int hB    = ncol * 16 + (lane & 15);   // this lane's h (B-layout N=lane%16)
    const int khalf = (lane >> 4) * 8;      // A-layout K offset for lane halves
    const int dB    = (lane >> 4) * 8;      // B-layout K offset

    // B fragments in registers: B[k=d][n=h] = W1[h][d]; elems 0..7 -> d = dB+[0..7]; 8..15 -> d = 16+dB+[0..7]
    v16h bfrag[4];
    #pragma unroll
    for (int kc = 0; kc < 4; ++kc) {
        const float* wp = W1 + hB * ND + kc * 32 + dB;
        v16h bf;
        #pragma unroll
        for (int e = 0; e < 16; ++e) bf[e] = (_Float16)wp[e < 8 ? e : e + 8];
        bfrag[kc] = bf;
    }
    const float vwl = Vw[hB];
    __syncthreads();

    const float w2 = s_w2h[hB];             // hoisted: constant per lane

    // ---- Phase 2: w1h GEMM (v_wmma_f32_16x16x32_f16) + fused tanh/V epilogue
    for (int mt = mgrp; mt < 13; mt += 2) {
        const _Float16* arow = &sA[(mt * 16 + (lane & 15)) * DP];
        // Load ALL four A fragments first (8x ds_load_b128, one wait) ...
        v16h afrag[4];
        #pragma unroll
        for (int kc = 0; kc < 4; ++kc) {
            // A 16-bit layout: elems 0..7 -> K = khalf+[0..7]; 8..15 -> K = 16+khalf+[0..7]
            half8 lo = *(const half8*)(arow + kc * 32 + khalf);
            half8 hi = *(const half8*)(arow + kc * 32 + 16 + khalf);
            afrag[kc] = __builtin_shufflevector(lo, hi,
                          0,1,2,3,4,5,6,7,8,9,10,11,12,13,14,15);
        }
        // ... then issue the 4 WMMAs back-to-back
        v8f c = {};
        #pragma unroll
        for (int kc = 0; kc < 4; ++kc)
            c = __builtin_amdgcn_wmma_f32_16x16x32_f16(
                    false, afrag[kc], false, bfrag[kc], (short)0, c, false, false);

        // C layout: lane group g = lane>>4; VGPR r -> (t = mt*16 + g*8 + r, h = hB)
        float part[8];
        #pragma unroll
        for (int r = 0; r < 8; ++r)
            part[r] = fast_tanh(c[r] + w2) * vwl;
        // reduce over 16 h's (lanes sharing a row): masks 1..8 stay inside each half
        #pragma unroll
        for (int m = 1; m < 16; m <<= 1) {
            #pragma unroll
            for (int r = 0; r < 8; ++r)
                part[r] += __shfl_xor(part[r], m, 32);
        }
        if ((lane & 15) == 0) {
            int tb = mt * 16 + (lane >> 4) * 8;
            #pragma unroll
            for (int r = 0; r < 8; ++r)
                s_attp[ncol][tb + r] = part[r];
        }
    }
    __syncthreads();

    // ---- Phase 3: masked softmax over T (wave 0; deterministic fixed-order sums)
    if (wid == 0) {
        const int* mrow = mask + (size_t)bs * NT;
        const float vb = Vb[0];
        float vals[7];
        float lmax = -3.0e38f;
        #pragma unroll
        for (int i = 0; i < 7; ++i) {
            int t = lane + i * 32;
            float a = -3.0e38f;
            if (t < NT) {
                a = vb + s_attp[0][t] + s_attp[1][t] + s_attp[2][t] + s_attp[3][t];
                if (mrow[t] == 0) a = -1.0e10f;
            }
            vals[i] = a;
            lmax = fmaxf(lmax, a);
        }
        #pragma unroll
        for (int m = 16; m >= 1; m >>= 1) lmax = fmaxf(lmax, __shfl_xor(lmax, m, 32));
        float lsum = 0.f;
        #pragma unroll
        for (int i = 0; i < 7; ++i) {
            int t = lane + i * 32;
            float e = (t < NT) ? fast_exp(vals[i] - lmax) : 0.f;
            vals[i] = e;
            lsum += e;
        }
        #pragma unroll
        for (int m = 16; m >= 1; m >>= 1) lsum += __shfl_xor(lsum, m, 32);
        float inv = __builtin_amdgcn_rcpf(lsum);
        #pragma unroll
        for (int i = 0; i < 7; ++i) {
            int t = lane + i * 32;
            if (t < NT) s_att[t] = vals[i] * inv;
        }
    }
    __syncthreads();

    // ---- Phase 4: out[d] = sum_t att[t] * input[t][d]; all 256 threads, split T
    {
        int d  = tid & (ND - 1);
        int hf = tid >> 7;                  // 0 or 1 -> t in [0,100) or [100,200)
        float acc = 0.f;
        int t0 = hf * (NT / 2);
        for (int t = t0; t < t0 + NT / 2; ++t)
            acc += s_att[t] * (float)sA[t * DP + d];
        s_attp[hf][d] = acc;                // reuse scratch (synced below)
    }
    __syncthreads();
    if (tid < ND)
        out[(size_t)bs * ND + tid] = s_attp[0][tid] + s_attp[1][tid];
}

extern "C" void kernel_launch(void* const* d_in, const int* in_sizes, int n_in,
                              void* d_out, int out_size, void* d_ws, size_t ws_size,
                              hipStream_t stream) {
    const float* inp  = (const float*)d_in[0];
    const float* z    = (const float*)d_in[1];
    const int*   mask = (const int*)  d_in[2];
    const float* W1   = (const float*)d_in[3];
    const float* W2w  = (const float*)d_in[4];
    const float* W2b  = (const float*)d_in[5];
    const float* Vw   = (const float*)d_in[6];
    const float* Vb   = (const float*)d_in[7];
    float* out = (float*)d_out;

    m2o_att_kernel<<<NB * NS, 256, 0, stream>>>(inp, z, mask, W1, W2w, W2b, Vw, Vb, out);
}
